// RoutedCausalLM_16707422781875
// MI455X (gfx1250) — compile-verified, matrix operand order checked by reading, not run on hardware
//
#include <hip/hip_runtime.h>

typedef __attribute__((ext_vector_type(16))) __bf16        v16bf;
typedef __attribute__((ext_vector_type(8)))  float         v8f;
typedef __attribute__((ext_vector_type(4)))  unsigned int  u32x4;
typedef __attribute__((ext_vector_type(4)))  float         f32x4;

struct Q2 { u32x4 lo, hi; };

constexpr int BATCH = 32;
constexpr int SEQ   = 1024;
constexpr int DIN   = 1024;
constexpr int DOUT  = 4096;
constexpr int RANK  = 16;
constexpr int NA    = 8;
constexpr int M     = BATCH * SEQ;            // 32768 rows
constexpr float SCALING = 32.0f / 16.0f;      // 2.0

// ---------------- helpers ----------------

__device__ __forceinline__ unsigned short f2bf(float f) {
  __bf16 h = (__bf16)f;
  return __builtin_bit_cast(unsigned short, h);
}

#if __has_builtin(__builtin_amdgcn_cvt_pk_bf16_f32)
typedef __attribute__((ext_vector_type(2))) __bf16 v2bf;
__device__ __forceinline__ unsigned pk2bf(float a, float b) {
  v2bf r = __builtin_amdgcn_cvt_pk_bf16_f32(a, b);   // one v_cvt_pk_bf16_f32
  return __builtin_bit_cast(unsigned, r);
}
#else
__device__ __forceinline__ unsigned pk2bf(float a, float b) {
  return (unsigned)f2bf(a) | ((unsigned)f2bf(b) << 16);
}
#endif

__device__ __forceinline__ v16bf frag16(u32x4 lo, u32x4 hi) {
  Q2 q; q.lo = lo; q.hi = hi;
  return __builtin_bit_cast(v16bf, q);
}

// CDNA5 async copy: 16B global -> LDS, tracked by ASYNCcnt
__device__ __forceinline__ void async16(unsigned lds_off, const unsigned short* g) {
  asm volatile("global_load_async_to_lds_b128 %0, %1, off"
               :: "v"(lds_off), "v"(g) : "memory");
}
__device__ __forceinline__ void wait_async_le8() {
  asm volatile("s_wait_asynccnt 0x8" ::: "memory");
}
__device__ __forceinline__ void wait_async_0() {
  asm volatile("s_wait_asynccnt 0x0" ::: "memory");
}

// ---------------- kernel 1: x f32 -> bf16 ----------------
__global__ __launch_bounds__(256) void k_cvt_x(const float* __restrict__ x,
                                               unsigned short* __restrict__ xb) {
  const long long i = (long long)blockIdx.x * 256 + threadIdx.x;   // exact grid
  const f32x4 a = ((const f32x4*)x)[2 * i];
  const f32x4 b = ((const f32x4*)x)[2 * i + 1];
  u32x4 o;
  o.x = pk2bf(a.x, a.y);
  o.y = pk2bf(a.z, a.w);
  o.z = pk2bf(b.x, b.y);
  o.w = pk2bf(b.z, b.w);
  ((u32x4*)xb)[i] = o;
}

// ---------------- kernel 2: W[K][N] -> bf16 Wt[N][K] (tiled transpose) ----------------
__global__ __launch_bounds__(256) void k_tr_w(const float* __restrict__ W,
                                              unsigned short* __restrict__ wt) {
  __shared__ float t[32][33];
  const int tx = threadIdx.x, ty = threadIdx.y;
  const int n0 = blockIdx.x * 32, k0 = blockIdx.y * 32;
#pragma unroll
  for (int i = 0; i < 4; ++i)
    t[ty + i * 8][tx] = W[(size_t)(k0 + ty + i * 8) * DOUT + n0 + tx];
  __syncthreads();
#pragma unroll
  for (int i = 0; i < 4; ++i)
    wt[(size_t)(n0 + ty + i * 8) * DIN + k0 + tx] = f2bf(t[tx][ty + i * 8]);
}

// ---------------- kernel 3: per-adapter Bt[a][n][32] (k>=16 zero) ----------------
__global__ __launch_bounds__(256) void k_bt(const float* __restrict__ lb,
                                            unsigned short* __restrict__ bt) {
  const int idx = blockIdx.x * 256 + threadIdx.x;   // < NA*DOUT*32 = 1048576
  const int k = idx & 31;
  const int n = (idx >> 5) & (DOUT - 1);
  const int a = idx >> 17;                          // DOUT*32 = 2^17
  unsigned short v = 0;
  if (k < RANK) v = f2bf(lb[((size_t)a * RANK + k) * DOUT + n]);
  bt[idx] = v;
}

// ---------------- kernel 4: lr[m][32] = SCALING * xb @ A[id]  (WMMA, N = RANK) ----------------
// reads the pre-converted bf16 activations -> no converts in the hot loop
__global__ __launch_bounds__(256) void k_lowrank(const unsigned short* __restrict__ xb,
                                                 const int* __restrict__ ids,
                                                 const float* __restrict__ la,
                                                 unsigned short* __restrict__ lr) {
  constexpr int LDA = 1032;  // padded stride (shorts): conflict-free ds_load_b128
  __shared__ __align__(16) unsigned short sAt[16 * LDA];

  const int tid = threadIdx.x;
  const int aid = ids[blockIdx.x >> 3];             // 128 rows/block, 1024 rows/sample
  const float* A = la + (size_t)aid * DIN * RANK;   // A[k][r]

  for (int i = tid; i < DIN * RANK; i += 256)       // transpose A -> sAt[r][k] (bf16)
    sAt[(i & 15) * LDA + (i >> 4)] = f2bf(A[i]);
  __syncthreads();

  const int wave = tid >> 5, lane = tid & 31, half = lane >> 4, l16 = lane & 15;
  const int mBase = blockIdx.x * 128 + wave * 16;
  const unsigned short* xr = xb + (size_t)(mBase + l16) * DIN;
  const unsigned short* atr = &sAt[l16 * LDA];

  v8f acc = {};
  for (int k0 = 0; k0 < DIN; k0 += 32) {
    v16bf av = frag16(*(const u32x4*)(xr + k0 + half * 8),
                      *(const u32x4*)(xr + k0 + 16 + half * 8));
    v16bf bv = frag16(*(const u32x4*)(atr + k0 + half * 8),
                      *(const u32x4*)(atr + k0 + 16 + half * 8));
    acc = __builtin_amdgcn_wmma_f32_16x16x32_bf16(false, av, false, bv,
                                                  (short)0, acc, false, false);
  }
#pragma unroll
  for (int r = 0; r < 8; ++r) {
    const size_t m = (size_t)mBase + r + 8 * half;
    lr[m * 32 + l16]      = f2bf(acc[r] * SCALING);
    lr[m * 32 + 16 + l16] = 0;                      // zero-pad K tail
  }
}

// ---------------- kernel 5: fused GEMM + LoRA delta + bias ----------------
// grid: x = N-tiles (32), y = M-tiles (256)  ->  x panel stays hot in L2 across the
// 32 N-blocks of each M-tile; W (8.4 MB bf16) is L2-resident across M-tiles.
// block 128x128, 8 waves (2M x 4N), wave tile 64x32. K-stage 64, double-buffered
// LDS fed by CDNA5 async-to-LDS copies (ASYNCcnt), issued two stages ahead.
constexpr int KT  = 64;   // K per stage (2 WMMA K-steps)
constexpr int LDT = 72;   // padded LDS row stride in shorts (conflict-free b128 reads)

__device__ __forceinline__ void issue_tile(const unsigned short* gX, const unsigned short* gW,
                                           unsigned ldsX, unsigned ldsW, int k0) {
#pragma unroll
  for (int i = 0; i < 4; ++i) {
    async16(ldsX + (unsigned)(i * 32 * LDT * 2), gX + k0 + (size_t)i * 32 * DIN);
    async16(ldsW + (unsigned)(i * 32 * LDT * 2), gW + k0 + (size_t)i * 32 * DIN);
  }
}

__global__ __launch_bounds__(256) void k_gemm(const unsigned short* __restrict__ xb,
                                              const unsigned short* __restrict__ wt,
                                              const unsigned short* __restrict__ lr,
                                              const unsigned short* __restrict__ bt,
                                              const float* __restrict__ bias,
                                              const int* __restrict__ ids,
                                              float* __restrict__ out) {
  __shared__ __align__(16) unsigned short sX[2][128 * LDT];   // 2 x 18 KB
  __shared__ __align__(16) unsigned short sW[2][128 * LDT];   // 2 x 18 KB

  const int tid = threadIdx.x;
  const int wave = tid >> 5, lane = tid & 31, half = lane >> 4, l16 = lane & 15;
  const int nBlock = blockIdx.x * 128, mBlock = blockIdx.y * 128;   // swapped grid
  const int waveM = (wave >> 2) * 64, waveN = (wave & 3) * 32;
  const int aid = ids[blockIdx.y >> 3];

  // prefetch LoRA-tail operands + bias while the K loop runs
  const unsigned short* lr0 = lr + (size_t)(mBlock + waveM + l16) * 32;
  const unsigned short* bt0 = bt + ((size_t)aid * DOUT + nBlock + waveN + l16) * 32;
  __builtin_prefetch(lr0, 0, 3);
  __builtin_prefetch(bt0, 0, 3);
  __builtin_prefetch(bias + nBlock + waveN + l16, 0, 3);

  v8f acc[4][2] = {};

  // async staging map: 1024 16B-chunks per operand tile, 4 per thread (rows r0+32i)
  const int r0  = tid >> 3;          // 0..31
  const int c0s = (tid & 7) * 8;     // 0..56 shorts
  const unsigned short* gX = xb + (size_t)(mBlock + r0) * DIN + c0s;
  const unsigned short* gW = wt + (size_t)(nBlock + r0) * DIN + c0s;
  unsigned ldsX[2], ldsW[2];
#pragma unroll
  for (int b = 0; b < 2; ++b) {
    ldsX[b] = (unsigned)(size_t)&sX[b][r0 * LDT + c0s];
    ldsW[b] = (unsigned)(size_t)&sW[b][r0 * LDT + c0s];
  }

  constexpr int NT = DIN / KT;       // 16 stages
  issue_tile(gX, gW, ldsX[0], ldsW[0], 0);
  issue_tile(gX, gW, ldsX[1], ldsW[1], KT);

  for (int t = 0; t < NT; ++t) {
    if (t + 1 < NT) wait_async_le8();   // stage t complete (stage t+1 may be in flight)
    else            wait_async_0();
    __syncthreads();

    const unsigned short* bufX = sX[t & 1];
    const unsigned short* bufW = sW[t & 1];
#pragma unroll
    for (int kk = 0; kk < KT; kk += 32) {
      v16bf a[4], b[2];
#pragma unroll
      for (int i = 0; i < 4; ++i) {
        const unsigned short* rp = &bufX[(waveM + i * 16 + l16) * LDT + kk];
        a[i] = frag16(*(const u32x4*)(rp + half * 8), *(const u32x4*)(rp + 16 + half * 8));
      }
#pragma unroll
      for (int j = 0; j < 2; ++j) {
        const unsigned short* rp = &bufW[(waveN + j * 16 + l16) * LDT + kk];
        b[j] = frag16(*(const u32x4*)(rp + half * 8), *(const u32x4*)(rp + 16 + half * 8));
      }
#pragma unroll
      for (int i = 0; i < 4; ++i)
#pragma unroll
        for (int j = 0; j < 2; ++j)
          acc[i][j] = __builtin_amdgcn_wmma_f32_16x16x32_bf16(false, a[i], false, b[j],
                                                              (short)0, acc[i][j], false, false);
    }
    __syncthreads();                  // everyone done reading buf (t&1)
    if (t + 2 < NT)
      issue_tile(gX, gW, ldsX[t & 1], ldsW[t & 1], (t + 2) * KT);
  }

  // LoRA tail: one extra K=32 WMMA per frag (upper 16 K zero-padded on both operands)
  {
    v16bf a[4], b[2];
#pragma unroll
    for (int i = 0; i < 4; ++i) {
      const unsigned short* rp = lr0 + (size_t)i * 16 * 32;
      a[i] = frag16(*(const u32x4*)(rp + half * 8), *(const u32x4*)(rp + 16 + half * 8));
    }
#pragma unroll
    for (int j = 0; j < 2; ++j) {
      const unsigned short* rp = bt0 + (size_t)j * 16 * 32;
      b[j] = frag16(*(const u32x4*)(rp + half * 8), *(const u32x4*)(rp + 16 + half * 8));
    }
#pragma unroll
    for (int i = 0; i < 4; ++i)
#pragma unroll
      for (int j = 0; j < 2; ++j)
        acc[i][j] = __builtin_amdgcn_wmma_f32_16x16x32_bf16(false, a[i], false, b[j],
                                                            (short)0, acc[i][j], false, false);
  }

  // epilogue: bias + non-temporal f32 store (write-once, keep L2 for x/W)
#pragma unroll
  for (int j = 0; j < 2; ++j) {
    const int n = nBlock + waveN + j * 16 + l16;
    const float bv = bias[n];
#pragma unroll
    for (int i = 0; i < 4; ++i) {
      const size_t mB = (size_t)mBlock + waveM + i * 16 + 8 * half;
#pragma unroll
      for (int r = 0; r < 8; ++r)
        __builtin_nontemporal_store(acc[i][j][r] + bv, &out[(mB + r) * DOUT + n]);
    }
  }
}

// ---------------- launch ----------------

extern "C" void kernel_launch(void* const* d_in, const int* in_sizes, int n_in,
                              void* d_out, int out_size, void* d_ws, size_t ws_size,
                              hipStream_t stream) {
  (void)in_sizes; (void)n_in; (void)out_size; (void)ws_size;
  const float* x    = (const float*)d_in[0];
  const int*   ids  = (const int*)d_in[1];
  const float* W    = (const float*)d_in[2];
  const float* bias = (const float*)d_in[3];
  const float* la   = (const float*)d_in[4];
  const float* lb   = (const float*)d_in[5];
  float* out = (float*)d_out;

  // workspace layout (all 16B-aligned): xb | wt | lr | bt  (~79.7 MB total)
  unsigned short* xb = (unsigned short*)d_ws;
  unsigned short* wt = xb + (size_t)M * DIN;
  unsigned short* lr = wt + (size_t)DOUT * DIN;
  unsigned short* bt = lr + (size_t)M * 32;

  k_cvt_x  <<<(M * DIN / 8) / 256, 256, 0, stream>>>(x, xb);
  k_tr_w   <<<dim3(DOUT / 32, DIN / 32), dim3(32, 8), 0, stream>>>(W, wt);
  k_bt     <<<(NA * DOUT * 32) / 256, 256, 0, stream>>>(lb, bt);
  k_lowrank<<<M / 128, 256, 0, stream>>>(xb, ids, la, lr);
  k_gemm   <<<dim3(DOUT / 128, M / 128), 256, 0, stream>>>(xb, wt, lr, bt, bias, ids, out);
}
